// MambaPlusPlus_layer_48103633715534
// MI455X (gfx1250) — matrix operand order repaired and checked by reading, not compiled, not run on hardware
//
#include <hip/hip_runtime.h>
#include <hip/hip_bf16.h>

// ---------------- problem constants ----------------
#define HB    16
#define DHD   64
#define HID   1024
#define EMB   1024
#define BSZ   4
#define LSEQ  2048
#define MROWS (BSZ * LSEQ)        // 8192
#define FFN   (4 * HID)           // 4096
#define EPSLN 1e-5f

#define NCH   16                  // scan chunks per sequence
#define CHL   (LSEQ / NCH)        // 128

// ---------------- WMMA tiling ----------------
#define BM  128
#define BN  128
#define BK  32
#define BKP 48                    // padded K stride for LDS B tile (96B rows, 32B aligned)

typedef __attribute__((ext_vector_type(16))) __bf16 v16bf;
typedef __attribute__((ext_vector_type(8)))  __bf16 v8bf;
typedef __attribute__((ext_vector_type(8)))  float  v8f;
typedef __attribute__((ext_vector_type(4)))  int    v4i;

// ---- gfx1250 async global->LDS staging (ASYNCcnt path), guarded ----
#if defined(__has_builtin)
#if __has_builtin(__builtin_amdgcn_global_load_async_to_lds_b128)
#define ASYNC_LDS 1
#endif
#endif

// builtin signature (from clang diagnostic): (AS1 v4i*, AS3 v4i*, int, int)
typedef __attribute__((address_space(1))) v4i* gptr_v4i;
typedef __attribute__((address_space(3))) v4i* lptr_v4i;

__device__ __forceinline__ void copy16B_to_lds(const __bf16* g, __bf16* l) {
#ifdef ASYNC_LDS
    __builtin_amdgcn_global_load_async_to_lds_b128((gptr_v4i)g, (lptr_v4i)l, 0, 0);
#else
    *(v8bf*)l = *(const v8bf*)g;
#endif
}

__device__ __forceinline__ void wait_lds_copies() {
#ifdef ASYNC_LDS
#if __has_builtin(__builtin_amdgcn_s_wait_asynccnt)
    __builtin_amdgcn_s_wait_asynccnt(0);
#else
    asm volatile("s_wait_asynccnt 0" ::: "memory");
#endif
#endif
}

__device__ __forceinline__ v16bf combine16(v8bf lo, v8bf hi) {
    v16bf r;
#pragma unroll
    for (int i = 0; i < 8; i++) { r[i] = lo[i]; r[i + 8] = hi[i]; }
    return r;
}

// =====================================================================
// Generic bf16 GEMM: C[M,N] = A[M,K] * BT[N,K]^T (+bias) (+epilogue)
// B is PRE-TRANSPOSED in the weight prepass -> all staging is b128.
// EPI 0: f32 out      EPI 1: tanh, f32 out
// EPI 2: exact GELU, bf16 out only
// EPI 3: +resid(f32), bf16 out only
// Tile: workgroup 128x128, 8 waves each 32x64 (2x4 16x16 tiles), K-step 32.
// =====================================================================
template <int EPI>
__global__ __launch_bounds__(256)
void gemm_bf16_kernel(const __bf16* __restrict__ A, const __bf16* __restrict__ BT,
                      const float* __restrict__ bias, const float* __restrict__ resid,
                      float* __restrict__ C, __bf16* __restrict__ C16,
                      int M, int N, int K)
{
    __shared__ __bf16 As[BM * BK];        // row-major A tile
    __shared__ __bf16 Bs[BN * BKP];       // Bs[n][k] (already transposed globally)

    const int tid  = threadIdx.x;
    const int lane = tid & 31;
    const int wave = tid >> 5;            // 0..7
    const int wrow = wave >> 1;           // 0..3 -> 32-row band
    const int wcol = wave & 1;            // 0..1 -> 64-col band
    const int ln16 = lane & 15;
    const int lhi  = lane >> 4;           // 0/1 (K-half select per ISA layout)

    const int n0 = blockIdx.x * BN;
    const int m0 = blockIdx.y * BM;

    const int ar = tid >> 1;              // A stage: row 0..127
    const int ak = (tid & 1) * 16;        //          k-chunk 0/16
    const int br = tid >> 1;              // B stage: n-row 0..127
    const int bk = (tid & 1) * 16;

    v8f acc[2][4];
#pragma unroll
    for (int s = 0; s < 2; s++)
#pragma unroll
        for (int t = 0; t < 4; t++)
#pragma unroll
            for (int i = 0; i < 8; i++) acc[s][t][i] = 0.0f;

    for (int k0 = 0; k0 < K; k0 += BK) {
        // ---- stage A: 128x32 bf16, 32B per thread (2x b128) ----
        {
            const __bf16* gp = A + (size_t)(m0 + ar) * K + k0 + ak;
            __bf16* lp = &As[ar * BK + ak];
            copy16B_to_lds(gp,     lp);
            copy16B_to_lds(gp + 8, lp + 8);
        }
        // ---- stage B: 128(n) x 32(k) from pre-transposed BT ----
        {
            const __bf16* gp = BT + (size_t)(n0 + br) * K + k0 + bk;
            __bf16* lp = &Bs[br * BKP + bk];
            copy16B_to_lds(gp,     lp);
            copy16B_to_lds(gp + 8, lp + 8);
        }
        wait_lds_copies();
        __syncthreads();

        if (k0 + BK < K) {   // global_prefetch_b8 of the next K tile
            __builtin_prefetch(A  + (size_t)(m0 + ar) * K + k0 + BK, 0, 0);
            __builtin_prefetch(BT + (size_t)(n0 + br) * K + k0 + BK, 0, 0);
        }

        // ---- fragments per CDNA5 lane layouts ----
        v16bf afrag[2], bfrag[4];
#pragma unroll
        for (int s = 0; s < 2; s++) {
            // A 16x32: lane holds row (ln16), K = {0..7,16..23} (+8 for hi lanes)
            const __bf16* p = &As[(wrow * 32 + s * 16 + ln16) * BK + lhi * 8];
            afrag[s] = combine16(*(const v8bf*)p, *(const v8bf*)(p + 16));
        }
#pragma unroll
        for (int t = 0; t < 4; t++) {
            // B 32x16: lane holds column (ln16), K = {0..7,16..23} (+8 for hi lanes)
            const __bf16* p = &Bs[(wcol * 64 + t * 16 + ln16) * BKP + lhi * 8];
            bfrag[t] = combine16(*(const v8bf*)p, *(const v8bf*)(p + 16));
        }
#pragma unroll
        for (int s = 0; s < 2; s++)
#pragma unroll
            for (int t = 0; t < 4; t++)
                acc[s][t] = __builtin_amdgcn_wmma_f32_16x16x32_bf16(
                    false, afrag[s], false, bfrag[t], (short)0, acc[s][t],
                    false, false);
        __syncthreads();
    }

    // ---- epilogue: C/D layout: VGPR i -> row i + 8*lhi, col ln16 ----
#pragma unroll
    for (int s = 0; s < 2; s++) {
#pragma unroll
        for (int t = 0; t < 4; t++) {
            const int col = n0 + wcol * 64 + t * 16 + ln16;
            const float bv = bias ? bias[col] : 0.0f;
#pragma unroll
            for (int i = 0; i < 8; i++) {
                const int row = m0 + wrow * 32 + s * 16 + lhi * 8 + i;
                const size_t o = (size_t)row * N + col;
                float v = acc[s][t][i] + bv;
                if (EPI == 1) v = tanhf(v);
                if (EPI == 2) v = 0.5f * v * (1.0f + erff(v * 0.70710678118f));
                if (EPI == 3) v += resid[o];
                if (EPI <= 1) C[o] = v;
                else          C16[o] = (__bf16)v;
            }
        }
    }
}

// =====================================================================
// Per-head WC GEMM: z[.,h,f] = (sum_d h16[.,h,d]*WC[h,d,f] + bC[h,f]) * d[.,h,f]
// WCT is pre-transposed per head: WCT[h][f][d].
// Workgroup = 128 rows x one head (N=64), 8 waves 32x32 each.
// =====================================================================
__global__ __launch_bounds__(256)
void head_wc_kernel(const __bf16* __restrict__ H16, const __bf16* __restrict__ WCT,
                    const float* __restrict__ bC, const float* __restrict__ Dv,
                    float* __restrict__ Z)
{
    __shared__ __bf16 As[128 * BK];
    __shared__ __bf16 Bs[64 * 80];        // Bs[f][d], padded row stride 80 (160B)

    const int h   = blockIdx.x;
    const int m0  = blockIdx.y * 128;
    const int tid = threadIdx.x;
    const int lane = tid & 31, wave = tid >> 5;
    const int wrow = wave >> 1;           // 0..3
    const int wcol = wave & 1;            // 0..1
    const int ln16 = lane & 15, lhi = lane >> 4;

    {   // stage whole 64x64 pre-transposed head matrix
        const __bf16* W = WCT + (size_t)h * DHD * DHD;
        const int f  = tid >> 2;           // 0..63
        const int kc = (tid & 3) * 16;     // 0,16,32,48
        copy16B_to_lds(W + f * DHD + kc,     &Bs[f * 80 + kc]);
        copy16B_to_lds(W + f * DHD + kc + 8, &Bs[f * 80 + kc + 8]);
    }

    v8f acc[2][2];
#pragma unroll
    for (int s = 0; s < 2; s++)
#pragma unroll
        for (int t = 0; t < 2; t++)
#pragma unroll
            for (int i = 0; i < 8; i++) acc[s][t][i] = 0.0f;

    for (int k0 = 0; k0 < DHD; k0 += BK) {
        const int r  = tid >> 1;
        const int kc = (tid & 1) * 16;
        const __bf16* gp = H16 + (size_t)(m0 + r) * HID + h * DHD + k0 + kc;
        __bf16* lp = &As[r * BK + kc];
        copy16B_to_lds(gp,     lp);
        copy16B_to_lds(gp + 8, lp + 8);
        wait_lds_copies();
        __syncthreads();

        v16bf afrag[2], bfrag[2];
#pragma unroll
        for (int s = 0; s < 2; s++) {
            const __bf16* p = &As[(wrow * 32 + s * 16 + ln16) * BK + lhi * 8];
            afrag[s] = combine16(*(const v8bf*)p, *(const v8bf*)(p + 16));
        }
#pragma unroll
        for (int t = 0; t < 2; t++) {
            const __bf16* p = &Bs[(wcol * 32 + t * 16 + ln16) * 80 + k0 + lhi * 8];
            bfrag[t] = combine16(*(const v8bf*)p, *(const v8bf*)(p + 16));
        }
#pragma unroll
        for (int s = 0; s < 2; s++)
#pragma unroll
            for (int t = 0; t < 2; t++)
                acc[s][t] = __builtin_amdgcn_wmma_f32_16x16x32_bf16(
                    false, afrag[s], false, bfrag[t], (short)0, acc[s][t],
                    false, false);
        __syncthreads();
    }

#pragma unroll
    for (int s = 0; s < 2; s++)
#pragma unroll
        for (int t = 0; t < 2; t++) {
            const int f = wcol * 32 + t * 16 + ln16;
            const float bc = bC[h * DHD + f];
#pragma unroll
            for (int i = 0; i < 8; i++) {
                const int row = m0 + wrow * 32 + s * 16 + lhi * 8 + i;
                const size_t o = (size_t)row * HID + h * DHD + f;
                Z[o] = (acc[s][t][i] + bc) * Dv[o];
            }
        }
}

// =====================================================================
// Chunked affine scan: h_t = a_t*h_{t-1} + b_t over L, per (batch, channel)
// =====================================================================
__global__ void scan_phase1_kernel(const float* __restrict__ Ag,
                                   const float* __restrict__ Bv,
                                   float* __restrict__ cA, float* __restrict__ cB)
{
    const int tid = blockIdx.x * blockDim.x + threadIdx.x;   // [b][chunk][ch]
    const int ch = tid & (HID - 1);
    const int c  = (tid >> 10) & (NCH - 1);
    const int b  = tid >> 14;
    const size_t base = ((size_t)b * LSEQ + (size_t)c * CHL) * HID + ch;
    float pa = 1.0f, pb = 0.0f;
    for (int t = 0; t < CHL; t++) {
        const float a = Ag[base + (size_t)t * HID];
        const float x = Bv[base + (size_t)t * HID];
        pa = a * pa;
        pb = a * pb + x;
    }
    cA[tid] = pa;
    cB[tid] = pb;
}

__global__ void scan_phase2_kernel(const float* __restrict__ cA,
                                   const float* __restrict__ cB,
                                   float* __restrict__ carry)
{
    const int tid = blockIdx.x * blockDim.x + threadIdx.x;   // [b][ch]
    const int ch = tid & (HID - 1);
    const int b  = tid >> 10;
    float hc = 0.0f;
    for (int c = 0; c < NCH; c++) {
        const size_t idx = ((size_t)b * NCH + c) * HID + ch;
        carry[idx] = hc;
        hc = cA[idx] * hc + cB[idx];
    }
}

__global__ void scan_phase3_kernel(const float* __restrict__ Ag,
                                   float* __restrict__ Bv,   // in-place -> h
                                   const float* __restrict__ carry)
{
    const int tid = blockIdx.x * blockDim.x + threadIdx.x;
    const int ch = tid & (HID - 1);
    const int c  = (tid >> 10) & (NCH - 1);
    const int b  = tid >> 14;
    const size_t base = ((size_t)b * LSEQ + (size_t)c * CHL) * HID + ch;
    float hst = carry[tid];
    for (int t = 0; t < CHL; t++) {
        const size_t o = base + (size_t)t * HID;
        hst = Ag[o] * hst + Bv[o];
        Bv[o] = hst;
    }
}

// =====================================================================
// Converters / weight repack (all B operands become [N][K] bf16)
// =====================================================================
__global__ void cvt_f32_bf16_kernel(const float* __restrict__ s,
                                    __bf16* __restrict__ d, size_t n)
{
    const size_t i = (size_t)blockIdx.x * blockDim.x + threadIdx.x;
    if (i < n) d[i] = (__bf16)s[i];
}

// W[H][E][DH] f32  ->  d[n = h*DH+dd][e]  (transposed [N][K], N=HID, K=EMB)
__global__ void permute_hed_T_kernel(const float* __restrict__ W,
                                     __bf16* __restrict__ d)
{
    const size_t i = (size_t)blockIdx.x * blockDim.x + threadIdx.x;
    if (i >= (size_t)HID * EMB) return;
    const int n  = (int)(i / EMB);
    const int e  = (int)(i % EMB);
    const int h  = n >> 6;
    const int dd = n & 63;
    d[i] = (__bf16)W[((size_t)h * EMB + e) * DHD + dd];
}

// src[K][N] f32 -> dst[n][k] bf16 (transpose-convert; write-coalesced)
__global__ void transpose_cvt_kernel(const float* __restrict__ src,
                                     __bf16* __restrict__ dst, int K, int N)
{
    const size_t i = (size_t)blockIdx.x * blockDim.x + threadIdx.x;
    if (i >= (size_t)K * N) return;
    const int n = (int)(i / K);
    const int k = (int)(i % K);
    dst[i] = (__bf16)src[(size_t)k * N + n];
}

// WC[H][D][F] f32 -> dst[h][f][d] bf16 (per-head transpose)
__global__ void wc_transpose_cvt_kernel(const float* __restrict__ src,
                                        __bf16* __restrict__ dst)
{
    const size_t i = (size_t)blockIdx.x * blockDim.x + threadIdx.x;
    if (i >= (size_t)HB * DHD * DHD) return;
    const int h  = (int)(i >> 12);
    const int f  = (int)((i >> 6) & 63);
    const int dd = (int)(i & 63);
    dst[i] = (__bf16)src[((size_t)h * DHD + dd) * DHD + f];
}

// =====================================================================
// u = z + LayerNorm(z)*g + b ; writes f32 and bf16 copies
// =====================================================================
__global__ __launch_bounds__(256)
void ln_kernel(const float* __restrict__ Z, const float* __restrict__ g,
               const float* __restrict__ bb, float* __restrict__ U,
               __bf16* __restrict__ U16)
{
    __shared__ float red[256];
    const int row = blockIdx.x;
    const float* z = Z + (size_t)row * HID;

    float x[4], s = 0.0f;
#pragma unroll
    for (int j = 0; j < 4; j++) { x[j] = z[threadIdx.x + j * 256]; s += x[j]; }
    red[threadIdx.x] = s;
    __syncthreads();
    for (int off = 128; off > 0; off >>= 1) {
        if (threadIdx.x < off) red[threadIdx.x] += red[threadIdx.x + off];
        __syncthreads();
    }
    const float mu = red[0] * (1.0f / HID);
    __syncthreads();

    float v2 = 0.0f;
#pragma unroll
    for (int j = 0; j < 4; j++) { const float t = x[j] - mu; v2 += t * t; }
    red[threadIdx.x] = v2;
    __syncthreads();
    for (int off = 128; off > 0; off >>= 1) {
        if (threadIdx.x < off) red[threadIdx.x] += red[threadIdx.x + off];
        __syncthreads();
    }
    const float rstd = rsqrtf(red[0] * (1.0f / HID) + EPSLN);

#pragma unroll
    for (int j = 0; j < 4; j++) {
        const int col = threadIdx.x + j * 256;
        const float u = x[j] + (x[j] - mu) * rstd * g[col] + bb[col];
        U[(size_t)row * HID + col] = u;
        U16[(size_t)row * HID + col] = (__bf16)u;
    }
}

// =====================================================================
// Host launcher
// =====================================================================
extern "C" void kernel_launch(void* const* d_in, const int* in_sizes, int n_in,
                              void* d_out, int out_size, void* d_ws, size_t ws_size,
                              hipStream_t stream)
{
    const float* emb    = (const float*)d_in[0];
    const float* Wa     = (const float*)d_in[1];
    const float* ba     = (const float*)d_in[2];
    const float* Wb     = (const float*)d_in[3];
    const float* bbv    = (const float*)d_in[4];
    const float* WD     = (const float*)d_in[5];
    const float* bD     = (const float*)d_in[6];
    const float* WC     = (const float*)d_in[7];
    const float* bC     = (const float*)d_in[8];
    const float* ffn1_w = (const float*)d_in[9];
    const float* ffn1_b = (const float*)d_in[10];
    const float* ffn2_w = (const float*)d_in[11];
    const float* ffn2_b = (const float*)d_in[12];
    const float* proj_w = (const float*)d_in[13];
    const float* proj_b = (const float*)d_in[14];
    const float* ln_g   = (const float*)d_in[15];
    const float* ln_b   = (const float*)d_in[16];

    char* ws = (char*)d_ws;
    size_t off = 0;
    auto alloc = [&](size_t bytes) -> size_t {
        size_t o = off; off += (bytes + 255) & ~(size_t)255; return o;
    };

    const size_t oEMB16 = alloc((size_t)MROWS * EMB * 2);   // emb bf16; reused as H16
    const size_t oWA16  = alloc((size_t)EMB * HID * 2);     // [N][K]
    const size_t oWB16  = alloc((size_t)EMB * HID * 2);
    const size_t oWD16  = alloc((size_t)EMB * HID * 2);
    const size_t oWC16  = alloc((size_t)HB * DHD * DHD * 2);// [h][f][d]
    const size_t oWF1   = alloc((size_t)HID * FFN * 2);     // [N=4096][K=1024]
    const size_t oWF2   = alloc((size_t)FFN * HID * 2);     // [N=1024][K=4096]
    const size_t oWP    = alloc((size_t)HID * HID * 2);     // [N][K]
    const size_t oAg    = alloc((size_t)MROWS * HID * 4);   // a gates; reused as Z
    const size_t oBv    = alloc((size_t)MROWS * HID * 4);   // b -> h (in place); reused as U16|V16
    const size_t oDv    = alloc((size_t)MROWS * HID * 4);   // d; reused as U (f32)
    const size_t oCA    = alloc((size_t)BSZ * NCH * HID * 4);
    const size_t oCB    = alloc((size_t)BSZ * NCH * HID * 4);
    const size_t oCY    = alloc((size_t)BSZ * NCH * HID * 4);
    const size_t oMID16 = alloc((size_t)MROWS * FFN * 2);
    (void)ws_size; (void)n_in; (void)in_sizes; (void)out_size;

    __bf16* EMB16 = (__bf16*)(ws + oEMB16);
    __bf16* H16   = (__bf16*)(ws + oEMB16);                 // alias (emb dead after projections)
    __bf16* WA16  = (__bf16*)(ws + oWA16);
    __bf16* WB16  = (__bf16*)(ws + oWB16);
    __bf16* WD16  = (__bf16*)(ws + oWD16);
    __bf16* WC16b = (__bf16*)(ws + oWC16);
    __bf16* WF1   = (__bf16*)(ws + oWF1);
    __bf16* WF2   = (__bf16*)(ws + oWF2);
    __bf16* WP    = (__bf16*)(ws + oWP);
    float*  Ag    = (float*)(ws + oAg);
    float*  Zf    = (float*)(ws + oAg);                      // alias (a dead after scan)
    float*  Bv    = (float*)(ws + oBv);
    __bf16* U16   = (__bf16*)(ws + oBv);                     // alias (h dead after ->bf16)
    __bf16* V16   = (__bf16*)(ws + oBv + (size_t)MROWS * HID * 2);
    float*  Dv    = (float*)(ws + oDv);
    float*  Uf    = (float*)(ws + oDv);                      // alias (d dead after head_wc)
    float*  cA    = (float*)(ws + oCA);
    float*  cB    = (float*)(ws + oCB);
    float*  carry = (float*)(ws + oCY);
    __bf16* MID16 = (__bf16*)(ws + oMID16);
    float*  out   = (float*)d_out;

    const int T = 256;
    auto gridN = [&](size_t n) { return (unsigned)((n + T - 1) / T); };

    // 1) precision conversion / weight repack (B operands -> [N][K] bf16)
    cvt_f32_bf16_kernel<<<gridN((size_t)MROWS * EMB), T, 0, stream>>>(emb, EMB16, (size_t)MROWS * EMB);
    permute_hed_T_kernel<<<gridN((size_t)HID * EMB), T, 0, stream>>>(Wa, WA16);
    permute_hed_T_kernel<<<gridN((size_t)HID * EMB), T, 0, stream>>>(Wb, WB16);
    permute_hed_T_kernel<<<gridN((size_t)HID * EMB), T, 0, stream>>>(WD, WD16);
    wc_transpose_cvt_kernel<<<gridN((size_t)HB * DHD * DHD), T, 0, stream>>>(WC, WC16b);
    transpose_cvt_kernel<<<gridN((size_t)HID * FFN), T, 0, stream>>>(ffn1_w, WF1, HID, FFN);
    transpose_cvt_kernel<<<gridN((size_t)FFN * HID), T, 0, stream>>>(ffn2_w, WF2, FFN, HID);
    transpose_cvt_kernel<<<gridN((size_t)HID * HID), T, 0, stream>>>(proj_w, WP, HID, HID);

    // 2) a/b/d projections (WMMA bf16)
    dim3 gP(HID / BN, MROWS / BM);
    gemm_bf16_kernel<1><<<gP, T, 0, stream>>>(EMB16, WA16, ba,  nullptr, Ag, nullptr, MROWS, HID, EMB);
    gemm_bf16_kernel<0><<<gP, T, 0, stream>>>(EMB16, WB16, bbv, nullptr, Bv, nullptr, MROWS, HID, EMB);
    gemm_bf16_kernel<0><<<gP, T, 0, stream>>>(EMB16, WD16, bD,  nullptr, Dv, nullptr, MROWS, HID, EMB);

    // 3) chunked affine scan over L (h written in place over Bv)
    scan_phase1_kernel<<<(BSZ * NCH * HID) / T, T, 0, stream>>>(Ag, Bv, cA, cB);
    scan_phase2_kernel<<<(BSZ * HID) / T, T, 0, stream>>>(cA, cB, carry);
    scan_phase3_kernel<<<(BSZ * NCH * HID) / T, T, 0, stream>>>(Ag, Bv, carry);

    // 4) h -> bf16 (into EMB16 slot)
    cvt_f32_bf16_kernel<<<gridN((size_t)MROWS * HID), T, 0, stream>>>(Bv, H16, (size_t)MROWS * HID);

    // 5) per-head WC GEMM, *d gate -> z (into Ag slot)
    head_wc_kernel<<<dim3(HB, MROWS / 128), T, 0, stream>>>(H16, WC16b, bC, Dv, Zf);

    // 6) u = z + LN(z); f32 into Dv slot, bf16 into Bv slot
    ln_kernel<<<MROWS, T, 0, stream>>>(Zf, ln_g, ln_b, Uf, U16);

    // 7) FFN1 + exact GELU -> bf16 mid
    gemm_bf16_kernel<2><<<dim3(FFN / BN, MROWS / BM), T, 0, stream>>>(
        U16, WF1, ffn1_b, nullptr, nullptr, MID16, MROWS, FFN, HID);

    // 8) FFN2 + residual u -> bf16 v
    gemm_bf16_kernel<3><<<dim3(HID / BN, MROWS / BM), T, 0, stream>>>(
        MID16, WF2, ffn2_b, Uf, nullptr, V16, MROWS, HID, FFN);

    // 9) projection -> f32 output
    gemm_bf16_kernel<0><<<dim3(HID / BN, MROWS / BM), T, 0, stream>>>(
        V16, WP, proj_b, nullptr, out, nullptr, MROWS, HID, HID);
}
